// DemoGPT_1967095022259
// MI455X (gfx1250) — matrix operand order repaired and checked by reading, not compiled, not run on hardware
//
#include <hip/hip_runtime.h>
#include <math.h>

typedef unsigned short u16;
typedef unsigned int   u32;
typedef __attribute__((ext_vector_type(16))) __bf16 v16bf;
typedef __attribute__((ext_vector_type(8)))  float  v8f;
typedef __attribute__((ext_vector_type(4)))  u32    v4u;
typedef __attribute__((ext_vector_type(8)))  u32    v8u;

struct __attribute__((aligned(16))) U4 { u32 x, y, z, w; };
union Frag { v16bf v; U4 q[2]; u16 s[16]; };

#define Bn   8
#define Tt   1024
#define Dd   768
#define Hh   12
#define HSs  64
#define Ll   6
#define F4   3072
#define NCc  10
#define MROWS (Bn*Tt)           // 8192
#define ATT_SCALE 0.125f        // 1/sqrt(64)
#define NEGBIG (-1.0e30f)

enum { OUT_BF16 = 0, OUT_GELU = 1, OUT_QK = 2, OUT_VT = 3, OUT_RESID = 4 };

// ---------- bf16 helpers (manual RNE, no libcalls) ----------
__device__ __forceinline__ u16 f2bf(float x) {
  u32 u = __float_as_uint(x);
  if ((u & 0x7F800000u) == 0x7F800000u) return (u16)(u >> 16); // inf/nan: truncate
  u32 r = u + 0x7FFFu + ((u >> 16) & 1u);
  return (u16)(r >> 16);
}

__device__ __forceinline__ v8f wmma_bf16(v16bf a, v16bf b, v8f c) {
  return __builtin_amdgcn_wmma_f32_16x16x32_bf16(false, a, false, b, (short)0, c,
                                                 false, false);
}

// A-fragment: 16x32 bf16, row-major source, lda in elements.
// lane -> row = lane&15 ; elements e<8 -> K = e + 8*half ; e>=8 -> K = e+8 + 8*half
__device__ __forceinline__ v16bf load_a(const u16* A, int lda, int row0, int k) {
  int lane = threadIdx.x & 31;
  int m = lane & 15, half = lane >> 4;
  const u16* p = A + (size_t)(row0 + m) * lda + k + 8 * half;
  Frag f;
  f.q[0] = *(const U4*)p;
  f.q[1] = *(const U4*)(p + 16);
  return f.v;
}

// B-fragment: 32x16 bf16 from pre-transposed weight Bt[N][K], ldb in elements.
// lane -> col = lane&15 ; elements e<8 -> K = e + 8*half ; e>=8 -> K = e+8 + 8*half
__device__ __forceinline__ v16bf load_b(const u16* Bt, int ldb, int col0, int k) {
  int lane = threadIdx.x & 31;
  int n = lane & 15, half = lane >> 4;
  const u16* p = Bt + (size_t)(col0 + n) * ldb + k + 8 * half;
  Frag f;
  f.q[0] = *(const U4*)p;
  f.q[1] = *(const U4*)(p + 16);
  return f.v;
}

// ---------- embedding ----------
__global__ void __launch_bounds__(256)
k_embed(const int* __restrict__ ids, const float* __restrict__ tok,
        const float* __restrict__ pos, float* __restrict__ x) {
  int bt = blockIdx.x;
  int t  = bt & (Tt - 1);
  int id = ids[bt];
  const float* tp = tok + (size_t)id * Dd;
  const float* pp = pos + (size_t)t * Dd;
  float* xp = x + (size_t)bt * Dd;
  for (int i = threadIdx.x; i < Dd; i += 256) xp[i] = tp[i] + pp[i];
}

// ---------- layernorm (one block per row) ----------
__global__ void __launch_bounds__(256)
k_ln(const float* __restrict__ x, const float* __restrict__ g,
     const float* __restrict__ b, void* __restrict__ out, int out_f32) {
  __shared__ float red[8];
  int row = blockIdx.x;
  const float* xr = x + (size_t)row * Dd;
  int lane = threadIdx.x & 31, wid = threadIdx.x >> 5;

  float s = 0.f;
  for (int i = threadIdx.x; i < Dd; i += 256) s += xr[i];
  for (int off = 16; off >= 1; off >>= 1) s += __shfl_xor(s, off, 32);
  if (lane == 0) red[wid] = s;
  __syncthreads();
  float tot = 0.f;
  #pragma unroll
  for (int i = 0; i < 8; i++) tot += red[i];
  float mean = tot * (1.f / Dd);
  __syncthreads();

  float v = 0.f;
  for (int i = threadIdx.x; i < Dd; i += 256) {
    float d = xr[i] - mean;
    v += d * d;
  }
  for (int off = 16; off >= 1; off >>= 1) v += __shfl_xor(v, off, 32);
  if (lane == 0) red[wid] = v;
  __syncthreads();
  float vt = 0.f;
  #pragma unroll
  for (int i = 0; i < 8; i++) vt += red[i];
  float inv = rsqrtf(vt * (1.f / Dd) + 1e-5f);

  for (int i = threadIdx.x; i < Dd; i += 256) {
    float val = (xr[i] - mean) * inv * g[i] + b[i];
    if (out_f32) ((float*)out)[(size_t)row * Dd + i] = val;
    else         ((u16*)out)[(size_t)row * Dd + i]   = f2bf(val);
  }
}

// ---------- weight converts ----------
// wq layer slice [H][D][HS] -> Wt[j=h*64+e][d], bf16
__global__ void __launch_bounds__(256)
k_cvt_qkv(const float* __restrict__ w, u16* __restrict__ wt) {
  int idx = blockIdx.x * 256 + threadIdx.x; // over 768*768
  int j = idx / Dd, d = idx - j * Dd;
  int h = j >> 6, e = j & 63;
  wt[idx] = f2bf(w[((size_t)h * Dd + d) * HSs + e]);
}

// generic transpose+convert: src f32 [R][C] -> dst bf16 [C][R]
__global__ void __launch_bounds__(256)
k_cvt_tr(const float* __restrict__ w, u16* __restrict__ wt, int R, int C) {
  int idx = blockIdx.x * 256 + threadIdx.x; // over R*C, dst-linear
  int c = idx / R, r = idx - c * R;
  wt[idx] = f2bf(w[(size_t)r * C + c]);
}

// ---------- GEMM: out(M,N) = A(M,K) x Wt(N,K)^T + bias, fused epilogues ----------
// block = 256 threads = 8 waves; WG tile 64x256; wave tile 32x64:
// 2 A-frags share 4 B-frags -> 8 WMMAs per 6 fragment loads.
// Ping-pong double buffering (x2 unrolled k-loop) -> no register copies.
__global__ void __launch_bounds__(256)
k_gemm(const u16* __restrict__ A, const u16* __restrict__ Bt,
       const float* __restrict__ bias, int N, int K, int mode,
       void* __restrict__ outp) {
  int wave = threadIdx.x >> 5;
  int lane = threadIdx.x & 31;
  int row0 = blockIdx.x * 64 + (wave >> 2) * 32;
  int col0 = blockIdx.y * 256 + (wave & 3) * 64;
  int half = lane >> 4, nl = lane & 15;

  v8f acc[2][4];
  v8f zf = {0.f, 0.f, 0.f, 0.f, 0.f, 0.f, 0.f, 0.f};
  #pragma unroll
  for (int rs = 0; rs < 2; rs++)
    #pragma unroll
    for (int j = 0; j < 4; j++) acc[rs][j] = zf;

  // two fragment sets, reloaded in place (ping-pong)
  v16bf a0[2], b0[4], a1[2], b1[4];
  a0[0] = load_a(A, K, row0, 0);
  a0[1] = load_a(A, K, row0 + 16, 0);
  #pragma unroll
  for (int j = 0; j < 4; j++) b0[j] = load_b(Bt, K, col0 + j * 16, 0);
  a1[0] = load_a(A, K, row0, 32);
  a1[1] = load_a(A, K, row0 + 16, 32);
  #pragma unroll
  for (int j = 0; j < 4; j++) b1[j] = load_b(Bt, K, col0 + j * 16, 32);

  for (int k = 64; k < K; k += 64) {   // K is a multiple of 64 (768 / 3072)
    __builtin_prefetch(Bt + (size_t)(col0 + nl) * K + k + 256, 0, 2);
    // consume set0 (chunk k-64), refill with chunk k
    #pragma unroll
    for (int j = 0; j < 4; j++) acc[0][j] = wmma_bf16(a0[0], b0[j], acc[0][j]);
    #pragma unroll
    for (int j = 0; j < 4; j++) acc[1][j] = wmma_bf16(a0[1], b0[j], acc[1][j]);
    a0[0] = load_a(A, K, row0, k);
    a0[1] = load_a(A, K, row0 + 16, k);
    #pragma unroll
    for (int j = 0; j < 4; j++) b0[j] = load_b(Bt, K, col0 + j * 16, k);
    // consume set1 (chunk k-32), refill with chunk k+32
    #pragma unroll
    for (int j = 0; j < 4; j++) acc[0][j] = wmma_bf16(a1[0], b1[j], acc[0][j]);
    #pragma unroll
    for (int j = 0; j < 4; j++) acc[1][j] = wmma_bf16(a1[1], b1[j], acc[1][j]);
    a1[0] = load_a(A, K, row0, k + 32);
    a1[1] = load_a(A, K, row0 + 16, k + 32);
    #pragma unroll
    for (int j = 0; j < 4; j++) b1[j] = load_b(Bt, K, col0 + j * 16, k + 32);
  }
  // drain both sets
  #pragma unroll
  for (int j = 0; j < 4; j++) acc[0][j] = wmma_bf16(a0[0], b0[j], acc[0][j]);
  #pragma unroll
  for (int j = 0; j < 4; j++) acc[1][j] = wmma_bf16(a0[1], b0[j], acc[1][j]);
  #pragma unroll
  for (int j = 0; j < 4; j++) acc[0][j] = wmma_bf16(a1[0], b1[j], acc[0][j]);
  #pragma unroll
  for (int j = 0; j < 4; j++) acc[1][j] = wmma_bf16(a1[1], b1[j], acc[1][j]);

  #pragma unroll
  for (int rs = 0; rs < 2; rs++) {
    int rbase = row0 + rs * 16;
    #pragma unroll
    for (int j = 0; j < 4; j++) {
      int n = col0 + j * 16 + nl;
      float bv = bias[n];
      #pragma unroll
      for (int r = 0; r < 8; r++) {
        int row = rbase + half * 8 + r;
        float val = acc[rs][j][r] + bv;
        if (mode == OUT_GELU) {
          val = 0.5f * val * (1.f + erff(val * 0.70710678118654752f));
          ((u16*)outp)[(size_t)row * N + n] = f2bf(val);
        } else if (mode == OUT_BF16) {
          ((u16*)outp)[(size_t)row * N + n] = f2bf(val);
        } else if (mode == OUT_QK) {
          int bb = row >> 10, t = row & (Tt - 1), hh = n >> 6, e = n & 63;
          ((u16*)outp)[(((size_t)(bb * Hh + hh) * Tt + t) << 6) + e] = f2bf(val);
        } else if (mode == OUT_VT) {
          int bb = row >> 10, t = row & (Tt - 1), hh = n >> 6, e = n & 63;
          ((u16*)outp)[((size_t)(bb * Hh + hh) * HSs + e) * Tt + t] = f2bf(val);
        } else { // OUT_RESID
          float* xo = (float*)outp;
          xo[(size_t)row * N + n] += val;
        }
      }
    }
  }
}

// ---------- flash attention: one wave handles a 16-row Q tile ----------
// Q,K: [B,H,T,64] bf16 ; Vt: [B,H,64,T] bf16 ; O: [B,T,768] bf16
// Q tile is DMA'd into LDS by the Tensor Data Mover (uniform 2D descriptor),
// then consumed as WMMA A-fragments via ds_load_b128.
__global__ void __launch_bounds__(32)
k_flash(const u16* __restrict__ Q, const u16* __restrict__ Km,
        const u16* __restrict__ Vt, u16* __restrict__ O) {
  __shared__ alignas(16) u16 lds_q[16 * 64]; // 2KB TDM destination (Q tile)
  __shared__ alignas(16) u16 lds_p[16 * 32]; // P tile staged for layout transpose
  int lane = threadIdx.x & 31;
  int half = lane >> 4, nl = lane & 15;

  int qt = blockIdx.x & 63;                 // T/16 tiles
  int h  = (blockIdx.x >> 6) % Hh;
  int b  = blockIdx.x / (64 * Hh);
  size_t bh = (size_t)(b * Hh + h);
  const u16* qptr = Q  + bh * Tt * HSs;
  const u16* kptr = Km + bh * Tt * HSs;
  const u16* vptr = Vt + bh * HSs * Tt;

  // ---- TDM: load 16x64 bf16 Q tile into LDS ----
  {
    size_t gaddr = (size_t)(const void*)(qptr + (size_t)qt * 16 * HSs);
    u32 ldsoff = (u32)(size_t)(void*)&lds_q[0];
    v4u g0;
    g0[0] = 1u;                                   // count=1, user descriptor
    g0[1] = ldsoff;                               // lds_addr
    g0[2] = (u32)gaddr;                           // global_addr[31:0]
    g0[3] = ((u32)(gaddr >> 32) & 0x01FFFFFFu)    // global_addr[56:32]
            | (2u << 30);                         // type=2 (image)
    v8u g1;
    g1[0] = 0x00010000u;           // wg_mask=0, data_size=1 (2 bytes)
    g1[1] = (64u << 16);           // tensor_dim0[15:0]=64 at bits 63:48
    g1[2] = (16u << 16);           // tensor_dim0 hi=0 | tensor_dim1[15:0]=16
    g1[3] = (64u << 16);           // tensor_dim1 hi=0 | tile_dim0=64
    g1[4] = 16u;                   // tile_dim1=16, tile_dim2=0
    g1[5] = 64u;                   // tensor_dim0_stride[31:0]=64
    g1[6] = 0u;                    // stride hi, dim1_stride lo
    g1[7] = 0u;                    // dim1_stride hi
    asm volatile("tensor_load_to_lds %0, %1" :: "s"(g0), "s"(g1) : "memory");
    __builtin_amdgcn_s_wait_tensorcnt(0);
  }

  // A-fragments of Q from LDS (ds_load_b128)
  Frag fq0, fq1;
  {
    const u16* p0 = &lds_q[nl * 64 + 8 * half];
    fq0.q[0] = *(const U4*)p0;
    fq0.q[1] = *(const U4*)(p0 + 16);
    const u16* p1 = &lds_q[nl * 64 + 32 + 8 * half];
    fq1.q[0] = *(const U4*)p1;
    fq1.q[1] = *(const U4*)(p1 + 16);
  }
  v16bf qa0 = fq0.v, qa1 = fq1.v;

  float mst[8], lst[8];
  v8f oac[4];
  v8f zf = {0.f, 0.f, 0.f, 0.f, 0.f, 0.f, 0.f, 0.f};
  #pragma unroll
  for (int f = 0; f < 4; f++) oac[f] = zf;
  #pragma unroll
  for (int r = 0; r < 8; r++) { mst[r] = NEGBIG; lst[r] = 0.f; }

  int jmax = (qt * 16 + 15) >> 5;           // last 32-key block (causal)
  for (int jb = 0; jb <= jmax; ++jb) {
    int key0 = jb * 32;
    // ---- S = Q x K^T for 32 keys (two 16-col C tiles) ----
    v16bf bk;
    bk = load_b(kptr, HSs, key0, 0);
    v8f s0 = wmma_bf16(qa0, bk, zf);
    bk = load_b(kptr, HSs, key0, 32);
    s0 = wmma_bf16(qa1, bk, s0);
    bk = load_b(kptr, HSs, key0 + 16, 0);
    v8f s1 = wmma_bf16(qa0, bk, zf);
    bk = load_b(kptr, HSs, key0 + 16, 32);
    s1 = wmma_bf16(qa1, bk, s1);

    // ---- causal mask + online softmax (row state is lane-local) ----
    float p0a[8], p1a[8];
    #pragma unroll
    for (int r = 0; r < 8; r++) {
      int row = qt * 16 + half * 8 + r;
      int c0 = key0 + nl, c1 = key0 + 16 + nl;
      float x0 = (c0 <= row) ? s0[r] * ATT_SCALE : NEGBIG;
      float x1 = (c1 <= row) ? s1[r] * ATT_SCALE : NEGBIG;
      float rm = fmaxf(x0, x1);
      rm = fmaxf(rm, __shfl_xor(rm, 1, 32));
      rm = fmaxf(rm, __shfl_xor(rm, 2, 32));
      rm = fmaxf(rm, __shfl_xor(rm, 4, 32));
      rm = fmaxf(rm, __shfl_xor(rm, 8, 32));
      float mnew = fmaxf(mst[r], rm);
      float sc = __expf(mst[r] - mnew);
      float p0 = __expf(x0 - mnew);
      float p1 = __expf(x1 - mnew);
      float rs = p0 + p1;
      rs += __shfl_xor(rs, 1, 32);
      rs += __shfl_xor(rs, 2, 32);
      rs += __shfl_xor(rs, 4, 32);
      rs += __shfl_xor(rs, 8, 32);
      lst[r] = lst[r] * sc + rs;
      mst[r] = mnew;
      #pragma unroll
      for (int f = 0; f < 4; f++) oac[f][r] = oac[f][r] * sc;
      p0a[r] = p0;
      p1a[r] = p1;
    }

    // ---- re-lay-out P (C-layout f32) -> A-layout bf16 via LDS ----
    #pragma unroll
    for (int r = 0; r < 8; r++) {
      int m = half * 8 + r;
      lds_p[m * 32 + nl]      = f2bf(p0a[r]);
      lds_p[m * 32 + 16 + nl] = f2bf(p1a[r]);
    }
    __syncthreads();
    Frag pa;
    const u16* lp = &lds_p[nl * 32 + 8 * half];
    pa.q[0] = *(const U4*)lp;
    pa.q[1] = *(const U4*)(lp + 16);
    __syncthreads();

    // ---- O += P x V (Vt rows are contiguous over keys -> clean B-frags) ----
    #pragma unroll
    for (int f = 0; f < 4; f++) {
      v16bf vb = load_b(vptr, Tt, f * 16, key0);
      oac[f] = wmma_bf16(pa.v, vb, oac[f]);
    }
  }

  // ---- normalize and scatter to O[b][t][h*64+e] ----
  #pragma unroll
  for (int f = 0; f < 4; f++) {
    #pragma unroll
    for (int r = 0; r < 8; r++) {
      int t = qt * 16 + half * 8 + r;
      int e = f * 16 + nl;
      float val = oac[f][r] / lst[r];
      O[((size_t)b * Tt + t) * Dd + h * HSs + e] = f2bf(val);
    }
  }
}

// ---------- mean pool over T (deterministic, no atomics) ----------
__global__ void __launch_bounds__(256)
k_pool(const float* __restrict__ lnf, float* __restrict__ pooled) {
  int b = blockIdx.x / 3;
  int d = (blockIdx.x % 3) * 256 + threadIdx.x;
  float s = 0.f;
  for (int t = 0; t < Tt; t++) s += lnf[((size_t)b * Tt + t) * Dd + d];
  pooled[b * Dd + d] = s * (1.f / Tt);
}

// ---------- classifier: one block per (b,c) ----------
__global__ void __launch_bounds__(256)
k_cls(const float* __restrict__ pooled, const float* __restrict__ wc,
      const float* __restrict__ bc, float* __restrict__ out) {
  __shared__ float red[8];
  int b = blockIdx.x / NCc, c = blockIdx.x % NCc;
  int lane = threadIdx.x & 31, wid = threadIdx.x >> 5;
  float s = 0.f;
  for (int i = threadIdx.x; i < Dd; i += 256)
    s += pooled[b * Dd + i] * wc[(size_t)i * NCc + c];
  for (int off = 16; off >= 1; off >>= 1) s += __shfl_xor(s, off, 32);
  if (lane == 0) red[wid] = s;
  __syncthreads();
  if (threadIdx.x == 0) {
    float t = 0.f;
    #pragma unroll
    for (int i = 0; i < 8; i++) t += red[i];
    out[b * NCc + c] = t + bc[c];
  }
}

// =====================================================================
extern "C" void kernel_launch(void* const* d_in, const int* in_sizes, int n_in,
                              void* d_out, int out_size, void* d_ws, size_t ws_size,
                              hipStream_t stream) {
  (void)in_sizes; (void)n_in; (void)out_size; (void)ws_size;
  const int*   ids  = (const int*)  d_in[0];
  const float* tok  = (const float*)d_in[1];
  const float* pos  = (const float*)d_in[2];
  const float* wq   = (const float*)d_in[3];
  const float* bq   = (const float*)d_in[4];
  const float* wk_  = (const float*)d_in[5];
  const float* bk   = (const float*)d_in[6];
  const float* wv   = (const float*)d_in[7];
  const float* bv   = (const float*)d_in[8];
  const float* wo   = (const float*)d_in[9];
  const float* bo   = (const float*)d_in[10];
  const float* ln1g = (const float*)d_in[11];
  const float* ln1b = (const float*)d_in[12];
  const float* w1   = (const float*)d_in[13];
  const float* b1   = (const float*)d_in[14];
  const float* w2   = (const float*)d_in[15];
  const float* b2   = (const float*)d_in[16];
  const float* ln2g = (const float*)d_in[17];
  const float* ln2b = (const float*)d_in[18];
  const float* lnfg = (const float*)d_in[19];
  const float* lnfb = (const float*)d_in[20];
  const float* wc   = (const float*)d_in[21];
  const float* bc   = (const float*)d_in[22];
  float* out = (float*)d_out;

  // workspace carve-up (all region sizes multiples of 256B)
  char* ws = (char*)d_ws;
  float* x    = (float*)ws; ws += (size_t)MROWS * Dd * 4;        // 25.2 MB residual
  u16*   h    = (u16*)  ws; ws += (size_t)MROWS * Dd * 2;        // LN out bf16
  u16*   qb   = (u16*)  ws; ws += (size_t)MROWS * Dd * 2;        // Q [B,H,T,64]
  u16*   kb   = (u16*)  ws; ws += (size_t)MROWS * Dd * 2;        // K [B,H,T,64]
  u16*   vtb  = (u16*)  ws; ws += (size_t)MROWS * Dd * 2;        // V^T [B,H,64,T]
  u16*   ob   = (u16*)  ws; ws += (size_t)MROWS * Dd * 2;        // attn out bf16
  char*  fraw = ws;         ws += (size_t)MROWS * F4 * 2;        // MLP mid bf16 / final-LN f32
  u16*   fb   = (u16*)fraw;
  float* lnf  = (float*)fraw;                                    // reused (25.2MB < 50.3MB)
  u16* wtq = (u16*)ws; ws += (size_t)Dd * Dd * 2;
  u16* wtk = (u16*)ws; ws += (size_t)Dd * Dd * 2;
  u16* wtv = (u16*)ws; ws += (size_t)Dd * Dd * 2;
  u16* wto = (u16*)ws; ws += (size_t)Dd * Dd * 2;
  u16* wt1 = (u16*)ws; ws += (size_t)Dd * F4 * 2;
  u16* wt2 = (u16*)ws; ws += (size_t)F4 * Dd * 2;
  float* pooled = (float*)ws; ws += (size_t)Bn * Dd * 4;

  dim3 g768(MROWS / 64, Dd / 256);    // (128,3)
  dim3 g3072(MROWS / 64, F4 / 256);   // (128,12)

  k_embed<<<MROWS, 256, 0, stream>>>(ids, tok, pos, x);

  for (int l = 0; l < Ll; l++) {
    // weight conversion for this layer
    k_cvt_qkv<<<(Dd * Dd) / 256, 256, 0, stream>>>(wq + (size_t)l * Hh * Dd * HSs, wtq);
    k_cvt_qkv<<<(Dd * Dd) / 256, 256, 0, stream>>>(wk_ + (size_t)l * Hh * Dd * HSs, wtk);
    k_cvt_qkv<<<(Dd * Dd) / 256, 256, 0, stream>>>(wv + (size_t)l * Hh * Dd * HSs, wtv);
    k_cvt_tr<<<(Dd * Dd) / 256, 256, 0, stream>>>(wo + (size_t)l * Dd * Dd, wto, Dd, Dd);
    k_cvt_tr<<<(Dd * F4) / 256, 256, 0, stream>>>(w1 + (size_t)l * Dd * F4, wt1, Dd, F4);
    k_cvt_tr<<<(F4 * Dd) / 256, 256, 0, stream>>>(w2 + (size_t)l * F4 * Dd, wt2, F4, Dd);

    // attention block
    k_ln<<<MROWS, 256, 0, stream>>>(x, ln1g + l * Dd, ln1b + l * Dd, h, 0);
    k_gemm<<<g768, 256, 0, stream>>>(h, wtq, bq + l * Dd, Dd, Dd, OUT_QK, qb);
    k_gemm<<<g768, 256, 0, stream>>>(h, wtk, bk + l * Dd, Dd, Dd, OUT_QK, kb);
    k_gemm<<<g768, 256, 0, stream>>>(h, wtv, bv + l * Dd, Dd, Dd, OUT_VT, vtb);
    k_flash<<<Bn * Hh * (Tt / 16), 32, 0, stream>>>(qb, kb, vtb, ob);
    k_gemm<<<g768, 256, 0, stream>>>(ob, wto, bo + l * Dd, Dd, Dd, OUT_RESID, x);

    // MLP block
    k_ln<<<MROWS, 256, 0, stream>>>(x, ln2g + l * Dd, ln2b + l * Dd, h, 0);
    k_gemm<<<g3072, 256, 0, stream>>>(h, wt1, b1 + l * F4, F4, Dd, OUT_GELU, fb);
    k_gemm<<<g768, 256, 0, stream>>>(fb, wt2, b2 + l * Dd, Dd, F4, OUT_RESID, x);
  }

  // final LN -> mean pool -> classifier
  k_ln<<<MROWS, 256, 0, stream>>>(x, lnfg, lnfb, lnf, 1);
  k_pool<<<Bn * 3, 256, 0, stream>>>(lnf, pooled);
  k_cls<<<Bn * NCc, 256, 0, stream>>>(pooled, wc, bc, out);
}
